// RGN_31885837206088
// MI455X (gfx1250) — matrix operand order repaired and checked by reading, not compiled, not run on hardware
//
#include <hip/hip_runtime.h>
#include <stdint.h>

// ---------------------------------------------------------------------------
// Graph network (2 layers) with zero-hidden GRUs:
//   GRU(x, h=0) = (1 - sigmoid(gi_z + bhh_z)) * tanh(gi_n + sigmoid(gi_r + bhh_r)*bhh_n)
//   where gi = x @ Wih.T + bih   (Whh unused since h=0)
// GEMMs run on v_wmma_f32_16x16x32_f16 (f16 in, f32 accumulate).
// ---------------------------------------------------------------------------

typedef __attribute__((ext_vector_type(16))) _Float16 v16h;
typedef __attribute__((ext_vector_type(8)))  float    v8f;

#define H64        64
#define NCOLS      192          // 3*H
#define NTILES     12           // 192/16
#define ROWS_PB    128          // rows per block
#define THREADS    256          // 8 waves
#define KPAD_MAX   160
#define KSKEW      8            // LDS row-stride pad (halves) = 16B, kills bank conflicts

__device__ __forceinline__ float sigmoid_f(float x) {
    return 1.0f / (1.0f + __expf(-x));
}
__device__ __forceinline__ float tanh_f(float x) {
    x = fminf(fmaxf(x, -15.0f), 15.0f);
    float t = __expf(2.0f * x);
    return (t - 1.0f) / (t + 1.0f);
}

// --------------------------- small utility kernels -------------------------

__global__ void k_zero(float* p, long n) {
    long i = (long)blockIdx.x * blockDim.x + threadIdx.x;
    if (i < n) p[i] = 0.0f;
}

__global__ void k_deg_count(const int* __restrict__ recv, float* __restrict__ deg, int E) {
    int i = blockIdx.x * blockDim.x + threadIdx.x;
    if (i < E) atomicAdd(&deg[recv[i]], 1.0f);
}

__global__ void k_inv_deg(const float* __restrict__ deg, float* __restrict__ inv, int n) {
    int i = blockIdx.x * blockDim.x + threadIdx.x;
    if (i < n) inv[i] = (deg[i] > 0.0f) ? (1.0f / deg[i]) : 0.0f;
}

// Pre-swizzle W ([192, K] row-major f32, used as B = W.T [K,192]) into the
// WMMA B-fragment layout, f16, zero-padded to Kpad (multiple of 32):
//   Bsw[ ((kt*12 + nt)*32 + lane)*16 + e ],  K_local=(lane>>4)*16+e, N_local=lane&15
__global__ void k_swizzle_w(const float* __restrict__ W, _Float16* __restrict__ out,
                            int K, int Kpad) {
    int t = blockIdx.x * blockDim.x + threadIdx.x;
    int total = Kpad * NCOLS;
    if (t >= total) return;
    int e  = t & 15;
    int l  = (t >> 4) & 31;
    int nt = (t >> 9) % NTILES;
    int kt = t / (512 * NTILES);
    int k  = kt * 32 + (l >> 4) * 16 + e;
    int n  = nt * 16 + (l & 15);
    float v = (k < K) ? W[n * K + k] : 0.0f;
    out[t] = (_Float16)v;
}

// agg[recv[e]] += he[e]  (mean-aggregation numerator)
__global__ void k_scatter_add(const float* __restrict__ he, const int* __restrict__ recv,
                              float* __restrict__ agg, int E) {
    long t = (long)blockIdx.x * blockDim.x + threadIdx.x;
    long total = (long)E * H64;
    if (t >= total) return;
    int e = (int)(t >> 6);
    int c = (int)(t & 63);
    atomicAdd(&agg[(long)recv[e] * H64 + c], he[t]);
}

// column sums of [R,64] matrix into sums[64]
__global__ void k_colsum(const float* __restrict__ m, float* __restrict__ sums, long R) {
    int c  = threadIdx.x & 63;
    int rl = threadIdx.x >> 6;                 // 0..3
    long r0   = (long)blockIdx.x * 512 + rl;
    long rend = (long)(blockIdx.x + 1) * 512;
    if (rend > R) rend = R;
    float s = 0.0f;
    for (long r = r0; r < rend; r += 4) s += m[r * H64 + c];
    atomicAdd(&sums[c], s);
}

// g_out = relu(gW @ [he_mean | hv_mean | g_prev] + gb), gW:[32,160]
__global__ void k_global_block(const float* __restrict__ sums, float invE, float invN,
                               const float* __restrict__ gprev,
                               const float* __restrict__ gW, const float* __restrict__ gb,
                               float* __restrict__ gout) {
    int j = threadIdx.x;
    if (j >= 32) return;
    const float* w = gW + j * 160;
    float acc = gb[j];
    for (int i = 0; i < 64; ++i) acc += w[i]       * (sums[i]      * invE);
    for (int i = 0; i < 64; ++i) acc += w[64 + i]  * (sums[64 + i] * invN);
    for (int i = 0; i < 32; ++i) acc += w[128 + i] * gprev[i];
    gout[j] = fmaxf(acc, 0.0f);
}

// --------------------------- fused GEMM + GRU kernel -----------------------
// Input row r = [ p0[r]*scale0[r] (d0) | p1[gather(r)] (d1) | gvec (dg) ], K = d0+d1+dg
// out[r] = GRU elementwise over gi = row @ W.T + bih   (W pre-swizzled in Bsw)
__global__ void __launch_bounds__(THREADS)
k_gru_gemm(const float* __restrict__ p0, int d0, const float* __restrict__ scale0,
           const float* __restrict__ p1, int d1, const int* __restrict__ gatherIdx,
           const float* __restrict__ gvec, int dg,
           const _Float16* __restrict__ Bsw, int KT, int Kpad,
           const float* __restrict__ bih, const float* __restrict__ bhh,
           float* __restrict__ out, long R)
{
    __shared__ _Float16 sIn[ROWS_PB * (KPAD_MAX + KSKEW)];

    const int  tid = threadIdx.x;
    const long rb  = (long)blockIdx.x * ROWS_PB;
    const int  KS  = Kpad + KSKEW;                // skewed LDS row stride (halves)
    const int  d01 = d0 + d1;
    const int  D   = d01 + dg;

    // ---- stage concatenated input rows into LDS as f16, zero-padded -------
    // 8 row-groups of 32 lanes; per-row gather index / scale hoisted.
    {
        const int lrow = tid >> 5;                // 0..7
        const int lcol = tid & 31;
        for (int i = lrow; i < ROWS_PB; i += 8) {
            long r = rb + i;
            bool valid = (r < R);
            long src = 0;
            float sc = 1.0f;
            if (valid) {
                src = gatherIdx ? (long)gatherIdx[r] : r;
                if (scale0) sc = scale0[r];
            }
            _Float16* dst = &sIn[i * KS];
            for (int c = lcol; c < Kpad; c += 32) {
                float v = 0.0f;
                if (valid && c < D) {
                    if (c < d0)        v = p0[r * d0 + c] * sc;
                    else if (c < d01)  v = p1[src * d1 + (c - d0)];
                    else               v = gvec[c - d01];
                }
                dst[c] = (_Float16)v;
            }
        }
    }
    __syncthreads();

    // ---- each wave computes a 16-row x 192-col tile -----------------------
    const int w    = tid >> 5;     // wave 0..7  -> rows rb + w*16 ..
    const int lane = tid & 31;
    const int hf   = lane >> 4;    // 0 / 1
    const int ml   = lane & 15;    // row (A) / column (C) within half

    v8f acc[NTILES];
    {
        v8f z;
        #pragma unroll
        for (int k = 0; k < 8; ++k) z[k] = 0.0f;
        #pragma unroll
        for (int i = 0; i < NTILES; ++i) acc[i] = z;
    }

    for (int kt = 0; kt < KT; ++kt) {
        // A fragment: 16x32 f16, contiguous K (lanes 0-15: K 0-15;
        // lanes 16-31: K 16-31), matches Bsw, pairs contiguous -> 8x b32 LDS reads.
        union { v16h v; uint32_t u[8]; } A;
        const uint32_t* rowp =
            (const uint32_t*)&sIn[(w * 16 + ml) * KS + kt * 32];
        #pragma unroll
        for (int vg = 0; vg < 8; ++vg)
            A.u[vg] = rowp[hf * 8 + vg];

        // Batch-load ALL 12 B fragments (24x b128 under one wait), then issue
        // the 12 WMMAs back-to-back without per-WMMA load stalls.
        const v16h* bp = (const v16h*)(Bsw + ((long)kt * NTILES * 32 + lane) * 16);
        v16h bf[NTILES];
        #pragma unroll
        for (int nt = 0; nt < NTILES; ++nt)
            bf[nt] = bp[nt * 32];                 // +1 KB per N-tile
        #pragma unroll
        for (int nt = 0; nt < NTILES; ++nt)
            acc[nt] = __builtin_amdgcn_wmma_f32_16x16x32_f16(
                false, A.v, false, bf[nt], (short)0, acc[nt], false, false);
    }

    // ---- fused GRU elementwise (h = 0) and store --------------------------
    // acc[0..3] = r-gate cols 0..63, acc[4..7] = z, acc[8..11] = n
    const long row0 = rb + w * 16;
    #pragma unroll
    for (int nt = 0; nt < 4; ++nt) {
        int j = nt * 16 + ml;                 // output column 0..63
        float bir = bih[j],        biz = bih[64 + j],  bin = bih[128 + j];
        float bhr = bhh[j],        bhz = bhh[64 + j],  bhn = bhh[128 + j];
        #pragma unroll
        for (int r = 0; r < 8; ++r) {
            long row = row0 + r + hf * 8;
            if (row < R) {
                float rg = sigmoid_f(acc[nt][r]     + bir + bhr);
                float zg = sigmoid_f(acc[4 + nt][r] + biz + bhz);
                float ng = tanh_f   (acc[8 + nt][r] + bin + rg * bhn);
                out[row * H64 + j] = (1.0f - zg) * ng;
            }
        }
    }
}

// --------------------------------- launcher --------------------------------

extern "C" void kernel_launch(void* const* d_in, const int* in_sizes, int n_in,
                              void* d_out, int out_size, void* d_ws, size_t ws_size,
                              hipStream_t stream) {
    (void)in_sizes; (void)n_in; (void)out_size; (void)ws_size;
    const int N = 50000, E = 800000;

    const float* x         = (const float*)d_in[0];
    const float* eF        = (const float*)d_in[1];
    const float* g0        = (const float*)d_in[2];
    const int*   senders   = (const int*)d_in[3];
    const int*   receivers = (const int*)d_in[4];
    const float* eW0  = (const float*)d_in[5];
    const float* ebih0= (const float*)d_in[7];
    const float* ebhh0= (const float*)d_in[8];
    const float* nW0  = (const float*)d_in[9];
    const float* nbih0= (const float*)d_in[11];
    const float* nbhh0= (const float*)d_in[12];
    const float* gW0  = (const float*)d_in[13];
    const float* gb0  = (const float*)d_in[14];
    const float* eW1  = (const float*)d_in[15];
    const float* ebih1= (const float*)d_in[17];
    const float* ebhh1= (const float*)d_in[18];
    const float* nW1  = (const float*)d_in[19];
    const float* nbih1= (const float*)d_in[21];
    const float* nbhh1= (const float*)d_in[22];
    const float* gW1  = (const float*)d_in[23];
    const float* gb1  = (const float*)d_in[24];

    float* he   = (float*)d_out;               // [E,64] (he0 then he1, in place)
    float* hv   = he + (long)E * H64;          // [N,64] (hv0 then hv1, in place)
    float* gout = hv + (long)N * H64;          // [32]

    // workspace carve
    char* ws = (char*)d_ws;
    float* deg  = (float*)ws;  ws += (size_t)N * sizeof(float);
    float* invd = (float*)ws;  ws += (size_t)N * sizeof(float);
    float* agg  = (float*)ws;  ws += (size_t)N * H64 * sizeof(float);
    float* gtmp = (float*)ws;  ws += 64 * sizeof(float);
    float* sums = (float*)ws;  ws += 128 * sizeof(float);
    uintptr_t pa = ((uintptr_t)ws + 255) & ~(uintptr_t)255;
    ws = (char*)pa;
    _Float16* e0sw = (_Float16*)ws; ws += (size_t) 96 * NCOLS * sizeof(_Float16);
    _Float16* n0sw = (_Float16*)ws; ws += (size_t)128 * NCOLS * sizeof(_Float16);
    _Float16* e1sw = (_Float16*)ws; ws += (size_t)160 * NCOLS * sizeof(_Float16);
    _Float16* n1sw = (_Float16*)ws; ws += (size_t)160 * NCOLS * sizeof(_Float16);

    auto cdiv = [](long a, long b) { return (int)((a + b - 1) / b); };

    // degree + inverse degree (once)
    k_zero<<<cdiv(N, 256), 256, 0, stream>>>(deg, N);
    k_deg_count<<<cdiv(E, 256), 256, 0, stream>>>(receivers, deg, E);
    k_inv_deg<<<cdiv(N, 256), 256, 0, stream>>>(deg, invd, N);

    // weight swizzles (B-fragment layout, f16)
    k_swizzle_w<<<cdiv( 96 * NCOLS, 256), 256, 0, stream>>>(eW0, e0sw,  80,  96);
    k_swizzle_w<<<cdiv(128 * NCOLS, 256), 256, 0, stream>>>(nW0, n0sw, 128, 128);
    k_swizzle_w<<<cdiv(160 * NCOLS, 256), 256, 0, stream>>>(eW1, e1sw, 160, 160);
    k_swizzle_w<<<cdiv(160 * NCOLS, 256), 256, 0, stream>>>(nW1, n1sw, 160, 160);

    // ---------------- layer 0 ----------------
    // edge: [e(16) | x[senders](32) | g(32)], K=80 -> Kpad 96 (3 K-tiles)
    k_gru_gemm<<<cdiv(E, ROWS_PB), THREADS, 0, stream>>>(
        eF, 16, nullptr, x, 32, senders, g0, 32,
        e0sw, 3, 96, ebih0, ebhh0, he, (long)E);
    // aggregate
    k_zero<<<cdiv((long)N * H64, 256), 256, 0, stream>>>(agg, (long)N * H64);
    k_scatter_add<<<cdiv((long)E * H64, 256), 256, 0, stream>>>(he, receivers, agg, E);
    // node: [agg*invd(64) | x(32) | g(32)], K=128 (4 K-tiles)
    k_gru_gemm<<<cdiv(N, ROWS_PB), THREADS, 0, stream>>>(
        agg, 64, invd, x, 32, nullptr, g0, 32,
        n0sw, 4, 128, nbih0, nbhh0, hv, (long)N);
    // global
    k_zero<<<1, 256, 0, stream>>>(sums, 128);
    k_colsum<<<cdiv(E, 512), 256, 0, stream>>>(he, sums, (long)E);
    k_colsum<<<cdiv(N, 512), 256, 0, stream>>>(hv, sums + 64, (long)N);
    k_global_block<<<1, 32, 0, stream>>>(sums, 1.0f / E, 1.0f / N, g0, gW0, gb0, gtmp);

    // ---------------- layer 1 ----------------
    // edge: [he0(64) | hv0[senders](64) | g1(32)], K=160 (5 K-tiles); he in place
    k_gru_gemm<<<cdiv(E, ROWS_PB), THREADS, 0, stream>>>(
        he, 64, nullptr, hv, 64, senders, gtmp, 32,
        e1sw, 5, 160, ebih1, ebhh1, he, (long)E);
    k_zero<<<cdiv((long)N * H64, 256), 256, 0, stream>>>(agg, (long)N * H64);
    k_scatter_add<<<cdiv((long)E * H64, 256), 256, 0, stream>>>(he, receivers, agg, E);
    // node: [agg*invd(64) | hv0(64) | g1(32)], K=160; hv in place
    k_gru_gemm<<<cdiv(N, ROWS_PB), THREADS, 0, stream>>>(
        agg, 64, invd, hv, 64, nullptr, gtmp, 32,
        n1sw, 5, 160, nbih1, nbhh1, hv, (long)N);
    // global -> final g output
    k_zero<<<1, 256, 0, stream>>>(sums, 128);
    k_colsum<<<cdiv(E, 512), 256, 0, stream>>>(he, sums, (long)E);
    k_colsum<<<cdiv(N, 512), 256, 0, stream>>>(hv, sums + 64, (long)N);
    k_global_block<<<1, 32, 0, stream>>>(sums, 1.0f / E, 1.0f / N, gtmp, gW1, gb1, gout);
}
